// TemporalAggregator_23545010717046
// MI455X (gfx1250) — matrix operand order repaired and checked by reading, not compiled, not run on hardware
//
#include <hip/hip_runtime.h>
#include <hip/hip_bf16.h>
#include <stdint.h>

// ---------------------------------------------------------------------------
// 2-layer bidirectional LSTM for MI455X (gfx1250, wave32, WMMA).
//  - Input projections: big GEMMs with v_wmma_f32_16x16x32_bf16 (f32 acc,
//    bias folded into C init), xp stored in C-fragment order for the scan.
//  - Recurrence: latency-bound; Whh^T fragments resident in VGPRs across the
//    whole T loop, h kept as bf16 in LDS in A-fragment order, cell state in
//    registers, xp ping-pong prefetched straight into the accumulators.
// ---------------------------------------------------------------------------

typedef __attribute__((ext_vector_type(16))) __bf16 v16bf;
typedef __attribute__((ext_vector_type(2)))  __bf16 v2bf;
typedef __attribute__((ext_vector_type(8)))  float  v8f;

#define HIDDEN 128
#define GATES  512
#define BATCH  64
#define GROW   516   // gates LDS row stride (floats) -> conflict-free exchange
#define HROW   136   // h LDS row stride (bf16), 16B-aligned rows + bank rotate

__device__ __forceinline__ __bf16 f2bf(float f) { return (__bf16)f; }

__device__ __forceinline__ v2bf f2bf2(float a, float b) {
#if __has_builtin(__builtin_amdgcn_cvt_pk_bf16_f32)
  return __builtin_amdgcn_cvt_pk_bf16_f32(a, b);
#else
  v2bf r; r[0] = (__bf16)a; r[1] = (__bf16)b; return r;
#endif
}

__device__ __forceinline__ float fast_rcp(float x) {
#if __has_builtin(__builtin_amdgcn_rcpf)
  return __builtin_amdgcn_rcpf(x);
#else
  return 1.0f / x;
#endif
}

__device__ __forceinline__ float fast_tanh(float x) {
#if __has_builtin(__builtin_amdgcn_tanhf)
  return __builtin_amdgcn_tanhf(x);
#else
  return tanhf(x);
#endif
}

__device__ __forceinline__ float fast_sigmoid(float x) {
  return fast_rcp(1.0f + __expf(-x));
}

union AFrag { uint4 q[2]; v16bf v; };
union CAcc  { float4 f4[2]; v8f v; float f[8]; };

// xp fragment-order index: [t][bblk(4)][wave(8)][j(4)][lane(32)][i(8)]
__device__ __forceinline__ size_t xp_idx(int t, int bblk, int wave, int j) {
  return ((((size_t)t * 4 + bblk) * 8 + wave) * 4 + j) * 256;
}

// ---------------------------------------------------------------------------
// xp = X @ Wih^T + bih + bhh, stored in C-fragment order (see xp_idx).
// X rows indexed [b][t] (x is [B,T,DIN], out1 is [B,T,256]).
// Grid: (T*B/16, 2=dir), block 256 (8 waves); wave w owns cols [64w, 64w+64).
// ---------------------------------------------------------------------------
template<int DIN>
__global__ __launch_bounds__(256, 1)
void xp_gemm(const float* __restrict__ X,
             const float* __restrict__ WihF, const float* __restrict__ WihB,
             const float* __restrict__ bihF, const float* __restrict__ bhhF,
             const float* __restrict__ bihB, const float* __restrict__ bhhB,
             float* __restrict__ xpF, float* __restrict__ xpB, int T)
{
  constexpr int KS = DIN / 32;
  const int dir = blockIdx.y;
  const float* W  = dir ? WihB : WihF;
  const float* bi = dir ? bihB : bihF;
  const float* bh = dir ? bhhB : bhhF;
  float*       xp = dir ? xpB  : xpF;

  const int lane  = threadIdx.x & 31;
  const int wave  = threadIdx.x >> 5;
  const int l15   = lane & 15;
  const int lhalf = lane >> 4;

  const int r0   = blockIdx.x * 16;      // output row tile base, r = t*64 + b
  const int t    = r0 >> 6;
  const int b0   = r0 & 63;
  const int bblk = b0 >> 4;

  // A fragments: 16 input rows (batch b0..b0+15 at timestep t).
  const float* ar = X + ((size_t)(b0 + l15) * T + t) * DIN + lhalf * 8;
  v16bf areg[KS];
  #pragma unroll
  for (int s = 0; s < KS; ++s) {
    #pragma unroll
    for (int e = 0; e < 4; ++e) {
      v2bf lo = f2bf2(ar[s * 32 + 2 * e],      ar[s * 32 + 2 * e + 1]);
      v2bf hi = f2bf2(ar[s * 32 + 16 + 2 * e], ar[s * 32 + 16 + 2 * e + 1]);
      areg[s][2 * e]     = lo[0]; areg[s][2 * e + 1] = lo[1];
      areg[s][8 + 2 * e] = hi[0]; areg[s][9 + 2 * e] = hi[1];
    }
  }

  #pragma unroll
  for (int j = 0; j < 4; ++j) {
    const int n = wave * 64 + j * 16 + l15;
    const float bv = bi[n] + bh[n];
    CAcc c;
    #pragma unroll
    for (int i = 0; i < 8; ++i) c.f[i] = bv;
    const float* wr = W + (size_t)n * DIN + lhalf * 8;
    #pragma unroll
    for (int s = 0; s < KS; ++s) {
      v16bf b;
      #pragma unroll
      for (int e = 0; e < 8; ++e) {
        v2bf p = f2bf2(wr[s * 32 + 2 * e + (e & 4) * 2], wr[s * 32 + 2 * e + (e & 4) * 2 + 1]);
        b[2 * e] = p[0]; b[2 * e + 1] = p[1];
      }
      c.v = __builtin_amdgcn_wmma_f32_16x16x32_bf16(false, areg[s], false, b,
                                                    (short)0, c.v, false, false);
    }
    float4* out = (float4*)(xp + xp_idx(t, bblk, wave, j) + lane * 8);
    out[0] = c.f4[0];
    out[1] = c.f4[1];
  }
}

// ---------------------------------------------------------------------------
// Persistent recurrence: WG owns 16 batch rows of one direction, scans T.
// MODE 0: write h each step to outSeq[b][t][colOff+j] (layer 0).
// MODE 1: write only final h to outFinal[b][colOff+j]  (layer 1).
// Grid: (4, 2=dir), block 256 (8 waves).
// ---------------------------------------------------------------------------
template<int MODE>
__global__ __launch_bounds__(256, 1)
void lstm_rec(const float* __restrict__ xpF, const float* __restrict__ xpB,
              const float* __restrict__ WhhF, const float* __restrict__ WhhB,
              float* __restrict__ outSeq, float* __restrict__ outFinal, int T)
{
  __shared__ float  gates[16 * GROW];     // ~33 KB
  __shared__ __bf16 hbuf[16 * HROW];      // ~4.3 KB; rows = batch, cols = hidden

  const int dir = blockIdx.y;
  const float* xp  = dir ? xpB  : xpF;
  const float* Whh = dir ? WhhB : WhhF;
  const int colOff = dir ? HIDDEN : 0;
  const int b0g    = blockIdx.x * 16;
  const int bblk   = blockIdx.x;

  const int lane  = threadIdx.x & 31;
  const int wave  = threadIdx.x >> 5;
  const int l15   = lane & 15;
  const int lhalf = lane >> 4;
  const int jew   = threadIdx.x & 127;    // elementwise hidden index
  const int mg    = (threadIdx.x >> 7) * 8;

  // Whh^T B fragments resident in VGPRs for the whole time loop.
  v16bf breg[4][4];
  #pragma unroll
  for (int j = 0; j < 4; ++j) {
    const int n = wave * 64 + j * 16 + l15;
    const float* wr = Whh + (size_t)n * HIDDEN + lhalf * 8;
    #pragma unroll
    for (int s = 0; s < 4; ++s)
      #pragma unroll
      for (int e = 0; e < 8; ++e) {
        v2bf p = f2bf2(wr[s * 32 + 2 * e + (e & 4) * 2], wr[s * 32 + 2 * e + (e & 4) * 2 + 1]);
        breg[j][s][2 * e] = p[0]; breg[j][s][2 * e + 1] = p[1];
      }
  }

  for (int idx = threadIdx.x; idx < 16 * HROW; idx += 256)
    hbuf[idx] = f2bf(0.0f);

  float cstate[8], hlast[8];
  #pragma unroll
  for (int i = 0; i < 8; ++i) { cstate[i] = 0.0f; hlast[i] = 0.0f; }
  __syncthreads();

  const int tstep = dir ? -1 : 1;
  int t = dir ? (T - 1) : 0;

  CAcc acc0[4], acc1[4];

  auto loadxp = [&](CAcc* dst, int tt) {
    #pragma unroll
    for (int j = 0; j < 4; ++j) {
      const float4* p = (const float4*)(xp + xp_idx(tt, bblk, wave, j) + lane * 8);
      dst[j].f4[0] = p[0];
      dst[j].f4[1] = p[1];
    }
  };

  // One full timestep: compute on `cur` (preloaded with xp(tt)), prefetch
  // xp(tt+tstep) into `nxt`.
  auto body = [&](CAcc* cur, CAcc* nxt, int tt, bool pref) {
    // A fragments from h (bf16 LDS) — two ds_load_b128 per k-step.
    AFrag a[4];
    const __bf16* hb = &hbuf[l15 * HROW + lhalf * 8];
    #pragma unroll
    for (int s = 0; s < 4; ++s) {
      a[s].q[0] = *(const uint4*)(hb + s * 32);
      a[s].q[1] = *(const uint4*)(hb + s * 32 + 16);
    }

    if (pref) loadxp(nxt, tt + tstep);   // hide HBM/L2 latency under the GEMM

    // gates += h @ Whh^T : 4 k-steps x 4 N tiles (chains of 4 per tile).
    #pragma unroll
    for (int s = 0; s < 4; ++s)
      #pragma unroll
      for (int j = 0; j < 4; ++j)
        cur[j].v = __builtin_amdgcn_wmma_f32_16x16x32_bf16(
            false, a[s].v, false, breg[j][s], (short)0, cur[j].v, false, false);

    // Spill gates to LDS (conflict-free C layout) for cross-wave combination.
    #pragma unroll
    for (int j = 0; j < 4; ++j) {
      const int n = wave * 64 + j * 16 + l15;
      float* g = &gates[(lhalf * 8) * GROW + n];
      #pragma unroll
      for (int i = 0; i < 8; ++i) g[i * GROW] = cur[j].f[i];
    }
    __syncthreads();

    // LSTM cell: each thread persistently owns 8 (m, j) cells.
    #pragma unroll
    for (int i = 0; i < 8; ++i) {
      const float* gr = &gates[(mg + i) * GROW];
      const float gi = gr[jew];
      const float gf = gr[HIDDEN + jew];
      const float gg = gr[2 * HIDDEN + jew];
      const float go = gr[3 * HIDDEN + jew];
      const float cs = fast_sigmoid(gf) * cstate[i] +
                       fast_sigmoid(gi) * fast_tanh(gg);
      cstate[i] = cs;
      const float h = fast_sigmoid(go) * fast_tanh(cs);
      hlast[i] = h;
      hbuf[(mg + i) * HROW + jew] = f2bf(h);
      if (MODE == 0)
        outSeq[(size_t)(b0g + mg + i) * T * 256 + (size_t)tt * 256 + colOff + jew] = h;
    }
    __syncthreads();   // h visible for next step; gates safe to overwrite
  };

  loadxp(acc0, t);
  for (int step = 0; step < T; step += 2) {
    body(acc0, acc1, t, step + 1 < T);
    t += tstep;
    if (step + 1 >= T) break;
    body(acc1, acc0, t, step + 2 < T);
    t += tstep;
  }

  if (MODE == 1) {
    #pragma unroll
    for (int i = 0; i < 8; ++i)
      outFinal[(size_t)(b0g + mg + i) * 256 + colOff + jew] = hlast[i];
  }
}

// ---------------------------------------------------------------------------
extern "C" void kernel_launch(void* const* d_in, const int* in_sizes, int n_in,
                              void* d_out, int out_size, void* d_ws, size_t ws_size,
                              hipStream_t stream) {
  (void)n_in; (void)out_size; (void)ws_size;

  const float* x     = (const float*)d_in[0];
  const float* Wih0f = (const float*)d_in[1];
  const float* Whh0f = (const float*)d_in[2];
  const float* bih0f = (const float*)d_in[3];
  const float* bhh0f = (const float*)d_in[4];
  const float* Wih0b = (const float*)d_in[5];
  const float* Whh0b = (const float*)d_in[6];
  const float* bih0b = (const float*)d_in[7];
  const float* bhh0b = (const float*)d_in[8];
  const float* Wih1f = (const float*)d_in[9];
  const float* Whh1f = (const float*)d_in[10];
  const float* bih1f = (const float*)d_in[11];
  const float* bhh1f = (const float*)d_in[12];
  const float* Wih1b = (const float*)d_in[13];
  const float* Whh1b = (const float*)d_in[14];
  const float* bih1b = (const float*)d_in[15];
  const float* bhh1b = (const float*)d_in[16];

  const int T = in_sizes[0] / (BATCH * 32);   // 2048

  // Workspace: xpF | xpB (T*64*512 f32 each) | out1 (B,T,256 f32) = 640 MB.
  float* xpF  = (float*)d_ws;
  float* xpB  = xpF + (size_t)T * BATCH * GATES;
  float* out1 = xpB + (size_t)T * BATCH * GATES;
  float* out  = (float*)d_out;

  dim3 blk(256);
  dim3 gXP((unsigned)((T * BATCH) / 16), 2);
  dim3 gRec(BATCH / 16, 2);

  // Layer 0: projections, then both-direction recurrence (writes out1).
  xp_gemm<32><<<gXP, blk, 0, stream>>>(x, Wih0f, Wih0b, bih0f, bhh0f,
                                       bih0b, bhh0b, xpF, xpB, T);
  lstm_rec<0><<<gRec, blk, 0, stream>>>(xpF, xpB, Whh0f, Whh0b, out1, nullptr, T);

  // Layer 1: projections from out1 (xp buffers reused), recurrence -> d_out.
  xp_gemm<256><<<gXP, blk, 0, stream>>>(out1, Wih1f, Wih1b, bih1f, bhh1f,
                                        bih1b, bhh1b, xpF, xpB, T);
  lstm_rec<1><<<gRec, blk, 0, stream>>>(xpF, xpB, Whh1f, Whh1b, nullptr, out, T);
}
